// StateActionVQTransformer_73735998538500
// MI455X (gfx1250) — matrix operand order repaired and checked by reading, not compiled, not run on hardware
//
#include <hip/hip_runtime.h>

typedef __attribute__((ext_vector_type(16))) _Float16 v16h;
typedef __attribute__((ext_vector_type(8)))  float    v8f;
typedef __attribute__((ext_vector_type(4)))  unsigned v4u;
typedef __attribute__((ext_vector_type(8)))  int      v8i;
typedef __attribute__((ext_vector_type(4)))  int      v4i;

// ---------------------------------------------------------------------------
// Fragment helpers (CDNA5 wave32 WMMA layouts, CDNA5 ISA §7.12.2)
// A (16x32 f16): lane m = lane&15, half=lane>>4; VGPR i holds K pair:
//   i<4 -> K = 2i + 8*half ;  i>=4 -> K = 16 + 2(i-4) + 8*half
// B (32x16 f16): lane n = lane&15, half=lane>>4; same K order as A (a_kbase)
// C/D (16x16 f32): element r, lane: row = r + 8*half, col = lane&15
// ---------------------------------------------------------------------------
__device__ __forceinline__ int a_kbase(int i, int hf) {
    int b = (i < 4) ? (2 * i) : (16 + 2 * (i - 4));
    return b + 8 * hf;
}

union HV {
    v16h v;
    unsigned u[8];
    _Float16 h[16];
};

union HV8 {
    int4 i;
    _Float16 h[8];
};

// ---------------------------------------------------------------------------
// Generic WMMA GEMM:  out = act(A[MxK] @ Bw[KxN] + bias (+ res))
// A, Bw f16 row-major. Writes f32 (outF) and/or f16 (outH).
// 256 threads = 8 waves in 2x4 grid; block tile 64x128; wave tile 32x32.
// Interior tiles: A tile staged via the Tensor Data Mover (TDM DMA, padded
// to the 40-half LDS pitch), B tile via unguarded b128 loads + transposed
// LDS scatter. Edge tiles: clamp+select (no exec-mask divergence).
// ---------------------------------------------------------------------------
template <int ACT, int RES>
__global__ __launch_bounds__(256) void gemm_kernel(
    const _Float16* __restrict__ A, const _Float16* __restrict__ Bw,
    const float* __restrict__ bias, const float* __restrict__ res,
    float* outF, _Float16* outH, int M, int N, int K) {
    __shared__ _Float16 sA[64][40];
    __shared__ _Float16 sBt[128][40];  // B tile transposed: [n][k]

    const int tid = threadIdx.x;
    const int m0 = blockIdx.y * 64, n0 = blockIdx.x * 128;
    const int lane = tid & 31, wave = tid >> 5;
    const int l16 = lane & 15, hf = lane >> 4;
    const int wm = wave >> 2, wn = wave & 3;  // 2 x 4 waves, 32x32 each

    v8f acc[2][2];
#pragma unroll
    for (int rg = 0; rg < 2; ++rg)
#pragma unroll
        for (int cg = 0; cg < 2; ++cg)
#pragma unroll
            for (int r = 0; r < 8; ++r) acc[rg][cg][r] = 0.f;

    const int ar = tid >> 2, ac0 = (tid & 3) * 8;    // A staging: 64 x 32
    const int bkr = tid >> 3, bc0 = (tid & 7) * 16;  // B staging: 32k x 128n
    const bool interior = (m0 + 64 <= M) && (n0 + 128 <= N) && ((K & 31) == 0);

    for (int k0 = 0; k0 < K; k0 += 32) {
        if (interior) {
            if (wave == 0) {
                // --- TDM: DMA the 64x32 f16 A tile into sA (padded pitch) ---
                unsigned lds0 = (unsigned)(size_t)(&sA[0][0]);
                unsigned long long ga =
                    (unsigned long long)(size_t)(A + (size_t)m0 * K + k0);
                v4u g0;
                g0[0] = 1u;  // count=1, is_restore=0, gather off
                g0[1] = lds0;
                g0[2] = (unsigned)(ga & 0xffffffffu);
                g0[3] = (unsigned)((ga >> 32) & 0x01ffffffu) | (2u << 30);  // type=2
                v8i g1;
                // data_size=2B(1), pad_enable, pad_interval=3 (16 DW=64B),
                // pad_amount=3 (4 DW=16B)
                g1[0] = (int)((1u << 16) | (1u << 20) | (3u << 22) | (3u << 25));
                g1[1] = (int)(32u << 16);  // tensor_dim0 = 32 (dim0 low in [63:48])
                g1[2] = (int)(64u << 16);  // tensor_dim1 = 64 (dim1 low in [95:80])
                g1[3] = (int)(32u << 16);  // tile_dim0 = 32 ([127:112])
                g1[4] = (int)64;           // tile_dim1 = 64 ([143:128])
                g1[5] = (int)K;            // tensor_dim0_stride = K
                g1[6] = 0;
                g1[7] = 0;
                v4i zz4;
                zz4[0] = 0; zz4[1] = 0; zz4[2] = 0; zz4[3] = 0;
                v8i zz8;
#pragma unroll
                for (int j = 0; j < 8; ++j) zz8[j] = 0;
                __builtin_amdgcn_tensor_load_to_lds(g0, g1, zz4, zz4, zz8, 0);
            }
            if (k0 + 32 < K)
                __builtin_prefetch(&Bw[(size_t)(k0 + 32 + bkr) * N + n0 + bc0], 0, 1);
            HV8 bv0, bv1;
            bv0.i = *(const int4*)(Bw + (size_t)(k0 + bkr) * N + n0 + bc0);
            bv1.i = *(const int4*)(Bw + (size_t)(k0 + bkr) * N + n0 + bc0 + 8);
#pragma unroll
            for (int j = 0; j < 8; ++j) sBt[bc0 + j][bkr] = bv0.h[j];
#pragma unroll
            for (int j = 0; j < 8; ++j) sBt[bc0 + 8 + j][bkr] = bv1.h[j];
            __builtin_amdgcn_s_wait_tensorcnt(0);
        } else {
            int arow = m0 + ar;
            int arc = arow < M ? arow : (M - 1);
            _Float16 ta[8];
#pragma unroll
            for (int j = 0; j < 8; ++j) {
                int kk = k0 + ac0 + j;
                int kkc = kk < K ? kk : (K - 1);
                ta[j] = A[(size_t)arc * K + kkc];
            }
#pragma unroll
            for (int j = 0; j < 8; ++j) {
                int kk = k0 + ac0 + j;
                sA[ar][ac0 + j] = (arow < M && kk < K) ? ta[j] : (_Float16)0.f;
            }
            int kk = k0 + bkr;
            int kkc = kk < K ? kk : (K - 1);
            _Float16 tb[16];
#pragma unroll
            for (int j = 0; j < 16; ++j) {
                int nn = n0 + bc0 + j;
                int nnc = nn < N ? nn : (N - 1);
                tb[j] = Bw[(size_t)kkc * N + nnc];
            }
#pragma unroll
            for (int j = 0; j < 16; ++j) {
                int nn = n0 + bc0 + j;
                sBt[bc0 + j][bkr] = (kk < K && nn < N) ? tb[j] : (_Float16)0.f;
            }
        }
        __syncthreads();

        HV a0, a1, b0, b1;
#pragma unroll
        for (int i = 0; i < 8; ++i) {
            a0.u[i] = *(const unsigned*)&sA[wm * 32 + l16][a_kbase(i, hf)];
            a1.u[i] = *(const unsigned*)&sA[wm * 32 + 16 + l16][a_kbase(i, hf)];
        }
#pragma unroll
        for (int v = 0; v < 8; ++v) {
            b0.u[v] = *(const unsigned*)&sBt[wn * 32 + l16][a_kbase(v, hf)];
            b1.u[v] = *(const unsigned*)&sBt[wn * 32 + 16 + l16][a_kbase(v, hf)];
        }
        acc[0][0] = __builtin_amdgcn_wmma_f32_16x16x32_f16(false, a0.v, false, b0.v,
                                                           (short)0, acc[0][0], false, false);
        acc[0][1] = __builtin_amdgcn_wmma_f32_16x16x32_f16(false, a0.v, false, b1.v,
                                                           (short)0, acc[0][1], false, false);
        acc[1][0] = __builtin_amdgcn_wmma_f32_16x16x32_f16(false, a1.v, false, b0.v,
                                                           (short)0, acc[1][0], false, false);
        acc[1][1] = __builtin_amdgcn_wmma_f32_16x16x32_f16(false, a1.v, false, b1.v,
                                                           (short)0, acc[1][1], false, false);
        __syncthreads();
    }

#pragma unroll
    for (int rg = 0; rg < 2; ++rg)
#pragma unroll
        for (int cg = 0; cg < 2; ++cg) {
            int gc = n0 + wn * 32 + cg * 16 + l16;
            if (gc >= N) continue;
            float bv = bias ? bias[gc] : 0.f;
#pragma unroll
            for (int r = 0; r < 8; ++r) {
                int gr = m0 + wm * 32 + rg * 16 + r + 8 * hf;
                if (gr >= M) continue;
                float val = acc[rg][cg][r] + bv;
                if (RES) val += res[(size_t)gr * N + gc];
                if (ACT) val = fmaxf(val, 0.f);
                if (outF) outF[(size_t)gr * N + gc] = val;
                if (outH) outH[(size_t)gr * N + gc] = (_Float16)val;
            }
        }
}

// ---------------------------------------------------------------------------
// 1-wave flash attention. Q/K/V/O are f16 (tokens x H*DH); head slice hoff.
// 16 q-rows per block, kv tiles of 32, online softmax, P@V via WMMA.
// All global loads are clamped+selected (no divergent-load exec juggling).
// ---------------------------------------------------------------------------
template <int DH>
__global__ __launch_bounds__(32) void attn_kernel(
    const _Float16* __restrict__ Qh, const _Float16* __restrict__ Kh,
    const _Float16* __restrict__ Vh, _Float16* __restrict__ Oh,
    int H, int Lq, int Lk, int causal) {
    constexpr int NC = DH / 32;  // K-chunks for Q@K^T
    constexpr int NO = DH / 16;  // output column groups
    const int Dm = H * DH;
    const int b = blockIdx.y / H, h = blockIdx.y % H;
    const int hoff = h * DH;
    const int qbase = b * Lq, kvbase = b * Lk;
    const int q0 = blockIdx.x * 16;
    const int lane = threadIdx.x & 31;
    const int l16 = lane & 15, hf = lane >> 4;
    const float scale = rsqrtf((float)DH);

    __shared__ _Float16 sP[16][40];

    HV qf[NC];
    {
        int tq = q0 + l16;
        if (tq >= Lq) tq = Lq - 1;
        const _Float16* qrow = Qh + (size_t)(qbase + tq) * Dm + hoff;
#pragma unroll
        for (int c = 0; c < NC; ++c)
#pragma unroll
            for (int i = 0; i < 8; ++i)
                qf[c].u[i] = *(const unsigned*)(qrow + c * 32 + a_kbase(i, hf));
    }

    float rmax[8], rsum[8];
    v8f oacc[NO];
#pragma unroll
    for (int r = 0; r < 8; ++r) { rmax[r] = -3.0e38f; rsum[r] = 0.f; }
#pragma unroll
    for (int c = 0; c < NO; ++c)
#pragma unroll
        for (int r = 0; r < 8; ++r) oacc[c][r] = 0.f;

    for (int jt = 0; jt < Lk; jt += 32) {
        v8f sc[2];
#pragma unroll
        for (int g = 0; g < 2; ++g)
#pragma unroll
            for (int r = 0; r < 8; ++r) sc[g][r] = 0.f;

#pragma unroll
        for (int c = 0; c < NC; ++c) {
#pragma unroll
            for (int g = 0; g < 2; ++g) {
                int tk = jt + g * 16 + l16;
                int tkc = tk < Lk ? tk : (Lk - 1);
                const _Float16* krow =
                    Kh + (size_t)(kvbase + tkc) * Dm + hoff + c * 32;
                HV kf;
#pragma unroll
                for (int v = 0; v < 8; ++v)
                    kf.u[v] = *(const unsigned*)(krow + a_kbase(v, hf));
                sc[g] = __builtin_amdgcn_wmma_f32_16x16x32_f16(
                    false, qf[c].v, false, kf.v, (short)0, sc[g], false, false);
            }
        }

        float corr[8];
#pragma unroll
        for (int r = 0; r < 8; ++r) {
            int qa = q0 + r + 8 * hf;
            float s0 = sc[0][r] * scale, s1 = sc[1][r] * scale;
            int k0a = jt + l16, k1a = jt + 16 + l16;
            if (k0a >= Lk || (causal && k0a > qa)) s0 = -3.0e38f;
            if (k1a >= Lk || (causal && k1a > qa)) s1 = -3.0e38f;
            float m = fmaxf(s0, s1);
#pragma unroll
            for (int msk = 1; msk < 16; msk <<= 1) m = fmaxf(m, __shfl_xor(m, msk));
            float nm = fmaxf(rmax[r], m);
            float cr = __expf(rmax[r] - nm);
            rmax[r] = nm;
            corr[r] = cr;
            float p0 = __expf(s0 - nm), p1 = __expf(s1 - nm);
            float ps = p0 + p1;
#pragma unroll
            for (int msk = 1; msk < 16; msk <<= 1) ps += __shfl_xor(ps, msk);
            rsum[r] = rsum[r] * cr + ps;
            sP[r + 8 * hf][l16] = (_Float16)p0;
            sP[r + 8 * hf][16 + l16] = (_Float16)p1;
        }
#pragma unroll
        for (int c = 0; c < NO; ++c)
#pragma unroll
            for (int r = 0; r < 8; ++r) oacc[c][r] *= corr[r];

        __syncthreads();
        HV pf;
#pragma unroll
        for (int i = 0; i < 8; ++i)
            pf.u[i] = *(const unsigned*)&sP[l16][a_kbase(i, hf)];
        __syncthreads();

#pragma unroll
        for (int c = 0; c < NO; ++c) {
            HV vf;
#pragma unroll
            for (int j = 0; j < 16; ++j) {
                int tk = jt + (j < 8 ? j : j + 8) + 8 * hf;
                int tkc = tk < Lk ? tk : (Lk - 1);
                _Float16 val =
                    Vh[(size_t)(kvbase + tkc) * Dm + hoff + c * 16 + l16];
                vf.h[j] = (tk < Lk) ? val : (_Float16)0.f;
            }
            oacc[c] = __builtin_amdgcn_wmma_f32_16x16x32_f16(
                false, pf.v, false, vf.v, (short)0, oacc[c], false, false);
        }
    }

#pragma unroll
    for (int c = 0; c < NO; ++c)
#pragma unroll
        for (int r = 0; r < 8; ++r) {
            int row = q0 + r + 8 * hf;
            if (row < Lq)
                Oh[(size_t)(qbase + row) * Dm + hoff + c * 16 + l16] =
                    (_Float16)(oacc[c][r] / rsum[r]);
        }
}

// ---------------------------------------------------------------------------
// Elementwise / reduction kernels
// ---------------------------------------------------------------------------
__global__ void cvt_kernel(const float* in, _Float16* out, int n) {
    int i = blockIdx.x * 256 + threadIdx.x;
    if (i < n) out[i] = (_Float16)in[i];
}

__global__ void patchify_kernel(const float* __restrict__ states,
                                _Float16* __restrict__ Ape) {
    int idx = blockIdx.x * 256 + threadIdx.x;  // 64*442*16 exactly
    int row = idx >> 4, e = idx & 15;
    int bt = row / 442, pr = row % 442;
    _Float16 v = (_Float16)0.f;
    if (pr > 0) {
        int p = pr - 1, py = p / 21, px = p % 21;
        int y = py * 4 + (e >> 2), x = px * 4 + (e & 3);
        v = (_Float16)states[(size_t)bt * 7056 + y * 84 + x];
    }
    Ape[idx] = v;
}

__global__ void set_cls_kernel(const float* cls, float* Xf, _Float16* Xh) {
    int bt = blockIdx.x, d = threadIdx.x;
    float v = cls[d];
    size_t o = (size_t)bt * 442 * 256 + d;
    Xf[o] = v;
    Xh[o] = (_Float16)v;
}

__global__ void seq_embed_kernel(const float* X, const int* actions,
                                 const float* emb, float* outF, _Float16* outH) {
    int bt = blockIdx.x, d = threadIdx.x;
    float v = X[(size_t)bt * 442 * 256 + d] + emb[(size_t)actions[bt] * 256 + d];
    outF[(size_t)bt * 256 + d] = v;
    outH[(size_t)bt * 256 + d] = (_Float16)v;
}

__global__ void ln_kernel(const float* __restrict__ X, const float* __restrict__ g,
                          const float* __restrict__ bb, float* __restrict__ outF,
                          _Float16* __restrict__ outH) {
    int row = blockIdx.x, tid = threadIdx.x;
    __shared__ float red[256];
    __shared__ float mv, vv;
    float v = X[(size_t)row * 256 + tid];
    red[tid] = v;
    __syncthreads();
    for (int s = 128; s > 0; s >>= 1) { if (tid < s) red[tid] += red[tid + s]; __syncthreads(); }
    if (tid == 0) mv = red[0] * (1.f / 256.f);
    __syncthreads();
    float dv = v - mv;
    red[tid] = dv * dv;
    __syncthreads();
    for (int s = 128; s > 0; s >>= 1) { if (tid < s) red[tid] += red[tid + s]; __syncthreads(); }
    if (tid == 0) vv = red[0] * (1.f / 256.f);
    __syncthreads();
    float o = dv * rsqrtf(vv + 1e-5f) * g[tid] + bb[tid];
    outF[(size_t)row * 256 + tid] = o;
    outH[(size_t)row * 256 + tid] = (_Float16)o;
}

__global__ void vq_kernel(const float* __restrict__ X, const float* __restrict__ cb,
                          float* __restrict__ zq, _Float16* __restrict__ zqh,
                          float* __restrict__ part) {
    int tok = blockIdx.x, tid = threadIdx.x;  // 128 threads, one per code
    __shared__ float xr[256];
    __shared__ float dl[128];
    __shared__ int il[128];
    xr[tid] = X[(size_t)tok * 256 + tid];
    xr[tid + 128] = X[(size_t)tok * 256 + tid + 128];
    __syncthreads();
    const float* c = cb + (size_t)tid * 256;
    float ds = 0.f;
    for (int d = 0; d < 256; ++d) { float df = xr[d] - c[d]; ds += df * df; }
    dl[tid] = ds;
    il[tid] = tid;
    __syncthreads();
    for (int s = 64; s > 0; s >>= 1) {
        if (tid < s) {
            if (dl[tid + s] < dl[tid] ||
                (dl[tid + s] == dl[tid] && il[tid + s] < il[tid])) {
                dl[tid] = dl[tid + s];
                il[tid] = il[tid + s];
            }
        }
        __syncthreads();
    }
    int idx = il[0];
    __syncthreads();
    const float* cq = cb + (size_t)idx * 256;
    float lsum = 0.f;
    for (int d = tid; d < 256; d += 128) {
        float z = cq[d];
        zq[(size_t)tok * 256 + d] = z;
        zqh[(size_t)tok * 256 + d] = (_Float16)z;
        float df = xr[d] - z;
        lsum += df * df;
    }
    dl[tid] = lsum;
    __syncthreads();
    for (int s = 64; s > 0; s >>= 1) { if (tid < s) dl[tid] += dl[tid + s]; __syncthreads(); }
    if (tid == 0) part[tok] = 2.0f * dl[0] / 16384.0f;  // c_loss + cb_loss
}

__global__ void build_cand_kernel(const float* __restrict__ OUTf,
                                  const float* __restrict__ Zq, float* __restrict__ Cf,
                                  _Float16* __restrict__ Ch, int L, int stepidx) {
    int row = blockIdx.x, d = threadIdx.x;
    int b = row / L, i = row % L;
    float v = (i == L - 1) ? Zq[(size_t)(b * 16 + stepidx) * 256 + d]
                           : OUTf[(size_t)(b * 16 + i) * 256 + d];
    Cf[(size_t)row * 256 + d] = v;
    Ch[(size_t)row * 256 + d] = (_Float16)v;
}

__global__ void save_step_kernel(const float* __restrict__ Cf, float* __restrict__ OUTf,
                                 _Float16* __restrict__ PREDh, int L, int t) {
    int row = blockIdx.x, d = threadIdx.x;
    int b = row / L, i = row % L;
    float v = Cf[(size_t)row * 256 + d];
    OUTf[(size_t)(b * 16 + i) * 256 + d] = v;
    if (i == L - 1) PREDh[(size_t)(b * 16 + t) * 256 + d] = (_Float16)v;
}

__global__ void rec_kernel(const float* __restrict__ rr, const float* __restrict__ states,
                           float* __restrict__ rec, float* __restrict__ part) {
    int idx = blockIdx.x * 256 + threadIdx.x;  // 451584 exactly
    int bt = idx / 7056, pix = idx % 7056;
    int y = pix / 84, x = pix % 84;
    float v = rr[bt * 16 + (y & 3) * 4 + (x & 3)];
    rec[idx] = v;
    float d = v - states[idx];
    d = d * d;
    __shared__ float red[256];
    red[threadIdx.x] = d;
    __syncthreads();
    for (int s = 128; s > 0; s >>= 1) {
        if (threadIdx.x < s) red[threadIdx.x] += red[threadIdx.x + s];
        __syncthreads();
    }
    if (threadIdx.x == 0) part[blockIdx.x] = red[0];
}

__global__ void loss_kernel(const float* rp, int nr, const float* vp, int nv, float* out) {
    __shared__ float red[256];
    int tid = threadIdx.x;
    float s = 0.f;
    for (int i = tid; i < nr; i += 256) s += rp[i];
    float sv = 0.f;
    for (int i = tid; i < nv; i += 256) sv += vp[i];
    red[tid] = s / 451584.0f + sv;
    __syncthreads();
    for (int st = 128; st > 0; st >>= 1) { if (tid < st) red[tid] += red[tid + st]; __syncthreads(); }
    if (tid == 0) out[0] = red[0];
}

// ---------------------------------------------------------------------------
// Host orchestration
// ---------------------------------------------------------------------------
extern "C" void kernel_launch(void* const* d_in, const int* in_sizes, int n_in,
                              void* d_out, int out_size, void* d_ws, size_t ws_size,
                              hipStream_t stream) {
    (void)in_sizes; (void)n_in; (void)out_size; (void)ws_size;

    // d_in order: states, actions, then params in jax tree (sorted-key) order.
    const float* states = (const float*)d_in[0];
    const int* actions = (const int*)d_in[1];
    const float* Wp = (const float*)d_in[2];
    const float* Wr = (const float*)d_in[3];
    const float* actEmb = (const float*)d_in[4];
    const float* aggP[16];
    for (int i = 0; i < 16; ++i) aggP[i] = (const float*)d_in[5 + i];
    const float* bp = (const float*)d_in[21];
    const float* br = (const float*)d_in[22];
    const float* cls = (const float*)d_in[23];
    const float* codebook = (const float*)d_in[24];
    const float* decP[26];
    for (int i = 0; i < 26; ++i) decP[i] = (const float*)d_in[25 + i];
    const float* encP[16];
    for (int i = 0; i < 16; ++i) encP[i] = (const float*)d_in[51 + i];
    // agg/enc key order: W1 W2 Wk Wo Wq Wv b1 b2 bk bo bq bv ln1_b ln1_s ln2_b ln2_s
    // dec key order: W1 W2 b1 b2 cross{Wk Wo Wq Wv bk bo bq bv}
    //                ln1_b ln1_s ln2_b ln2_s ln3_b ln3_s self{Wk Wo Wq Wv bk bo bq bv}

    char* base = (char*)d_ws;
    size_t off = 0;
    auto alloc = [&](size_t bytes) -> void* {
        off = (off + 255) & ~(size_t)255;
        void* p = base + off;
        off += bytes;
        return p;
    };

    const int MAGG = 64 * 442;  // 28288
    float* Xf = (float*)alloc((size_t)MAGG * 256 * 4);
    float* TMP = (float*)alloc((size_t)MAGG * 256 * 4);
    _Float16* Xh = (_Float16*)alloc((size_t)MAGG * 256 * 2);
    _Float16* Qb = (_Float16*)alloc((size_t)MAGG * 256 * 2);
    _Float16* Kb = (_Float16*)alloc((size_t)MAGG * 256 * 2);
    _Float16* Vb = (_Float16*)alloc((size_t)MAGG * 256 * 2);
    _Float16* Ab = (_Float16*)alloc((size_t)MAGG * 256 * 2);
    _Float16* Fh = (_Float16*)alloc((size_t)MAGG * 256 * 2);
    _Float16* Ape = (_Float16*)alloc((size_t)MAGG * 16 * 2);
    float* Zq = (float*)alloc(64 * 256 * 4);
    _Float16* Zqh = (_Float16*)alloc(64 * 256 * 2);
    _Float16 *KM[4], *VM[4];
    for (int l = 0; l < 4; ++l) {
        KM[l] = (_Float16*)alloc(64 * 256 * 2);
        VM[l] = (_Float16*)alloc(64 * 256 * 2);
    }
    float* OUTf = (float*)alloc(64 * 256 * 4);
    float* CANDf = (float*)alloc(64 * 256 * 4);
    _Float16* CANDh = (_Float16*)alloc(64 * 256 * 2);
    _Float16* PREDh = (_Float16*)alloc(64 * 256 * 2);
    float* rr = (float*)alloc(64 * 16 * 4);
    float* vqPart = (float*)alloc(64 * 4);
    float* recPart = (float*)alloc(1764 * 4);

    auto cvt = [&](const float* src, size_t n) -> _Float16* {
        _Float16* dst = (_Float16*)alloc(n * 2);
        cvt_kernel<<<dim3((unsigned)((n + 255) / 256)), dim3(256), 0, stream>>>(
            src, dst, (int)n);
        return dst;
    };
    auto gemm = [&](const _Float16* A, const _Float16* W, const float* bias,
                    const float* res, float* oF, _Float16* oH, int M, int N, int K,
                    int act) {
        dim3 g((N + 127) / 128, (M + 63) / 64), blk(256);
        if (res) {
            if (act) gemm_kernel<1, 1><<<g, blk, 0, stream>>>(A, W, bias, res, oF, oH, M, N, K);
            else gemm_kernel<0, 1><<<g, blk, 0, stream>>>(A, W, bias, res, oF, oH, M, N, K);
        } else {
            if (act) gemm_kernel<1, 0><<<g, blk, 0, stream>>>(A, W, bias, res, oF, oH, M, N, K);
            else gemm_kernel<0, 0><<<g, blk, 0, stream>>>(A, W, bias, res, oF, oH, M, N, K);
        }
    };
    auto attn = [&](const _Float16* q, const _Float16* k, const _Float16* v,
                    _Float16* o, int Bs, int H, int dh, int Lq, int Lk, int causal) {
        dim3 g((Lq + 15) / 16, Bs * H), blk(32);
        if (dh == 64) attn_kernel<64><<<g, blk, 0, stream>>>(q, k, v, o, H, Lq, Lk, causal);
        else attn_kernel<32><<<g, blk, 0, stream>>>(q, k, v, o, H, Lq, Lk, causal);
    };
    auto ln = [&](const float* x, const float* s, const float* b2, float* oF,
                  _Float16* oH, int M) {
        ln_kernel<<<dim3(M), dim3(256), 0, stream>>>(x, s, b2, oF, oH);
    };

    // f16 weight mirrors
    _Float16* WpH = cvt(Wp, 16 * 256);
    _Float16* WrH = cvt(Wr, 256 * 16);
    _Float16 *aW1[2], *aW2[2], *aWk[2], *aWo[2], *aWq[2], *aWv[2];
    for (int l = 0; l < 2; ++l) {
        aW1[l] = cvt(aggP[0] + (size_t)l * 65536, 65536);
        aW2[l] = cvt(aggP[1] + (size_t)l * 65536, 65536);
        aWk[l] = cvt(aggP[2] + (size_t)l * 65536, 65536);
        aWo[l] = cvt(aggP[3] + (size_t)l * 65536, 65536);
        aWq[l] = cvt(aggP[4] + (size_t)l * 65536, 65536);
        aWv[l] = cvt(aggP[5] + (size_t)l * 65536, 65536);
    }
    _Float16 *eW1[4], *eW2[4], *eWk[4], *eWo[4], *eWq[4], *eWv[4];
    for (int l = 0; l < 4; ++l) {
        eW1[l] = cvt(encP[0] + (size_t)l * 131072, 131072);
        eW2[l] = cvt(encP[1] + (size_t)l * 131072, 131072);
        eWk[l] = cvt(encP[2] + (size_t)l * 65536, 65536);
        eWo[l] = cvt(encP[3] + (size_t)l * 65536, 65536);
        eWq[l] = cvt(encP[4] + (size_t)l * 65536, 65536);
        eWv[l] = cvt(encP[5] + (size_t)l * 65536, 65536);
    }
    _Float16 *dW1[4], *dW2[4], *dWkC[4], *dWoC[4], *dWqC[4], *dWvC[4];
    _Float16 *dWkS[4], *dWoS[4], *dWqS[4], *dWvS[4];
    for (int l = 0; l < 4; ++l) {
        dW1[l] = cvt(decP[0] + (size_t)l * 131072, 131072);
        dW2[l] = cvt(decP[1] + (size_t)l * 131072, 131072);
        dWkC[l] = cvt(decP[4] + (size_t)l * 65536, 65536);
        dWoC[l] = cvt(decP[5] + (size_t)l * 65536, 65536);
        dWqC[l] = cvt(decP[6] + (size_t)l * 65536, 65536);
        dWvC[l] = cvt(decP[7] + (size_t)l * 65536, 65536);
        dWkS[l] = cvt(decP[18] + (size_t)l * 65536, 65536);
        dWoS[l] = cvt(decP[19] + (size_t)l * 65536, 65536);
        dWqS[l] = cvt(decP[20] + (size_t)l * 65536, 65536);
        dWvS[l] = cvt(decP[21] + (size_t)l * 65536, 65536);
    }

    auto enc_layer = [&](float* Xf_, _Float16* Xh_, int M, int Bs, int L, int H,
                         int dh, int ff, const _Float16* Wq_, const float* bq,
                         const _Float16* Wk_, const float* bk, const _Float16* Wv_,
                         const float* bv, const _Float16* Wo_, const float* bo,
                         const _Float16* W1_, const float* b1, const _Float16* W2_,
                         const float* b2, const float* l1s, const float* l1b,
                         const float* l2s, const float* l2b) {
        gemm(Xh_, Wq_, bq, nullptr, nullptr, Qb, M, 256, 256, 0);
        gemm(Xh_, Wk_, bk, nullptr, nullptr, Kb, M, 256, 256, 0);
        gemm(Xh_, Wv_, bv, nullptr, nullptr, Vb, M, 256, 256, 0);
        attn(Qb, Kb, Vb, Ab, Bs, H, dh, L, L, 0);
        gemm(Ab, Wo_, bo, Xf_, TMP, nullptr, M, 256, 256, 0);
        ln(TMP, l1s, l1b, Xf_, Xh_, M);
        gemm(Xh_, W1_, b1, nullptr, nullptr, Fh, M, ff, 256, 1);
        gemm(Fh, W2_, b2, Xf_, TMP, nullptr, M, 256, ff, 0);
        ln(TMP, l2s, l2b, Xf_, Xh_, M);
    };

    // ---- PatchEmbedding + cls ----
    patchify_kernel<<<1768, 256, 0, stream>>>(states, Ape);
    gemm(Ape, WpH, bp, nullptr, Xf, Xh, MAGG, 256, 16, 0);
    set_cls_kernel<<<64, 256, 0, stream>>>(cls, Xf, Xh);

    // ---- PatchAggregator: 2 layers, 64 seqs of 442 tokens, h=4, dh=64, ff=256
    for (int l = 0; l < 2; ++l)
        enc_layer(Xf, Xh, MAGG, 64, 442, 4, 64, 256, aWq[l], aggP[10] + l * 256,
                  aWk[l], aggP[8] + l * 256, aWv[l], aggP[11] + l * 256, aWo[l],
                  aggP[9] + l * 256, aW1[l], aggP[6] + l * 256, aW2[l],
                  aggP[7] + l * 256, aggP[13] + l * 256, aggP[12] + l * 256,
                  aggP[15] + l * 256, aggP[14] + l * 256);

    // ---- f_agg + action embedding -> seq (64 tokens) ----
    seq_embed_kernel<<<64, 256, 0, stream>>>(Xf, actions, actEmb, Xf, Xh);

    // ---- SeqEncoder: 4 layers, 4 seqs of 16 tokens, h=8, dh=32, ff=512 ----
    for (int l = 0; l < 4; ++l)
        enc_layer(Xf, Xh, 64, 4, 16, 8, 32, 512, eWq[l], encP[10] + l * 256,
                  eWk[l], encP[8] + l * 256, eWv[l], encP[11] + l * 256, eWo[l],
                  encP[9] + l * 256, eW1[l], encP[6] + l * 512, eW2[l],
                  encP[7] + l * 256, encP[13] + l * 256, encP[12] + l * 256,
                  encP[15] + l * 256, encP[14] + l * 256);

    // ---- Vector quantizer (forward pass q == z_q) ----
    vq_kernel<<<64, 128, 0, stream>>>(Xf, codebook, Zq, Zqh, vqPart);

    // ---- Precompute cross-attention K/V of memory (= z_q) per decoder layer
    for (int l = 0; l < 4; ++l) {
        gemm(Zqh, dWkC[l], decP[8] + l * 256, nullptr, nullptr, KM[l], 64, 256, 256, 0);
        gemm(Zqh, dWvC[l], decP[11] + l * 256, nullptr, nullptr, VM[l], 64, 256, 256, 0);
    }

    // ---- Autoregressive decoder, 16 teacher-forced steps ----
    for (int t = 0; t < 16; ++t) {
        int L = t + 1, M = 4 * L, stepidx = (t == 0) ? 0 : (t - 1);
        build_cand_kernel<<<4 * L, 256, 0, stream>>>(OUTf, Zq, CANDf, CANDh, L, stepidx);
        for (int l = 0; l < 4; ++l) {
            // self attention (causal)
            gemm(CANDh, dWqS[l], decP[24] + l * 256, nullptr, nullptr, Qb, M, 256, 256, 0);
            gemm(CANDh, dWkS[l], decP[22] + l * 256, nullptr, nullptr, Kb, M, 256, 256, 0);
            gemm(CANDh, dWvS[l], decP[25] + l * 256, nullptr, nullptr, Vb, M, 256, 256, 0);
            attn(Qb, Kb, Vb, Ab, 4, 8, 32, L, L, 1);
            gemm(Ab, dWoS[l], decP[23] + l * 256, CANDf, TMP, nullptr, M, 256, 256, 0);
            ln(TMP, decP[13] + l * 256, decP[12] + l * 256, CANDf, CANDh, M);
            // cross attention
            gemm(CANDh, dWqC[l], decP[10] + l * 256, nullptr, nullptr, Qb, M, 256, 256, 0);
            attn(Qb, KM[l], VM[l], Ab, 4, 8, 32, L, 16, 0);
            gemm(Ab, dWoC[l], decP[9] + l * 256, CANDf, TMP, nullptr, M, 256, 256, 0);
            ln(TMP, decP[15] + l * 256, decP[14] + l * 256, CANDf, CANDh, M);
            // FFN
            gemm(CANDh, dW1[l], decP[2] + l * 512, nullptr, nullptr, Fh, M, 512, 256, 1);
            gemm(Fh, dW2[l], decP[3] + l * 256, CANDf, TMP, nullptr, M, 256, 512, 0);
            ln(TMP, decP[17] + l * 256, decP[16] + l * 256, CANDf, CANDh, M);
        }
        save_step_kernel<<<4 * L, 256, 0, stream>>>(CANDf, OUTf, PREDh, L, t);
    }

    // ---- FrameDecoder (all patches share one 4x4 block per frame) + losses
    gemm(PREDh, WrH, br, nullptr, rr, nullptr, 64, 16, 256, 0);
    rec_kernel<<<1764, 256, 0, stream>>>(rr, states, (float*)d_out, recPart);
    loss_kernel<<<1, 256, 0, stream>>>(recPart, 1764, vqPart, 64,
                                       (float*)d_out + 451584);
}
